// DeletionLayer_66400194396169
// MI455X (gfx1250) — matrix operand order repaired and checked by reading, not compiled, not run on hardware
//
#include <hip/hip_runtime.h>

// ---------------------------------------------------------------------------
// DeletionLayer: out = mask ? x @ W : x     (N=131072, DIM=512, f32)
// bf16x3 split GEMM on v_wmma_f32_16x16x32_bf16; W pre-split/transposed once;
// per-chunk B staging via Tensor Data Mover (tensor_load_to_lds + TENSORcnt).
// ---------------------------------------------------------------------------

typedef __bf16 bf16;
typedef bf16  v16bf __attribute__((ext_vector_type(16)));
typedef float v8f   __attribute__((ext_vector_type(8)));
typedef unsigned int u32;
typedef u32 v4u __attribute__((ext_vector_type(4)));
typedef int  v4i __attribute__((ext_vector_type(4)));
typedef int  v8i __attribute__((ext_vector_type(8)));

#define DIM            512
#define WG_THREADS     256          // 8 waves of 32
#define ROWS_PER_WAVE  16
#define ROWS_PER_WG    128
#define NCHUNK         64           // output columns staged per LDS pass
#define NCHUNKS        (DIM / NCHUNK)
#define LDSB_ELEMS     (NCHUNK * DIM)   // 32768 bf16 per plane = 64 KB

#if defined(__has_builtin)
#if __has_builtin(__builtin_amdgcn_tensor_load_to_lds)
#define HAVE_TDM 1
#endif
#if __has_builtin(__builtin_amdgcn_s_wait_tensorcnt)
#define HAVE_TCNT 1
#endif
#endif

// ---------------------------------------------------------------------------
// Prep: split W (f32, [k][n]) into bf16 hi/lo planes transposed to [n][k].
// ---------------------------------------------------------------------------
__global__ void split_transpose_w(const float* __restrict__ W,
                                  bf16* __restrict__ whiT,
                                  bf16* __restrict__ wloT) {
    int idx = blockIdx.x * blockDim.x + threadIdx.x;   // over DIM*DIM, [n][k]
    int n = idx >> 9;
    int k = idx & (DIM - 1);
    float w = W[(size_t)k * DIM + n];
    bf16 h = (bf16)w;
    whiT[idx] = h;
    wloT[idx] = (bf16)(w - (float)h);
}

#if HAVE_TDM
// Issue one TDM descriptor copying a contiguous 64 KB block global -> LDS.
// D# group0: count=1, lds_addr, global_addr[56:0], type=2.
// D# group1: data_size=8B, 1-D tile: tile_dim0 = tensor_dim0 = stride = 8192.
__device__ __forceinline__ void tdm_load_64kb(u32 lds_off, const void* gptr) {
    unsigned long long ga = (unsigned long long)(uintptr_t)gptr;
    v4u g0;
    g0.x = 1u;                                          // count=1, no gather
    g0.y = lds_off;                                     // lds_addr (bytes)
    g0.z = (u32)(ga & 0xFFFFFFFFu);                     // global_addr[31:0]
    g0.w = (u32)((ga >> 32) & 0x01FFFFFFu) | (2u << 30);// addr[56:32], type=2
    v8i g1;
    g1[0] = (int)(3u << 16);         // workgroup_mask=0, data_size=3 (8B)
    g1[1] = (int)(0x2000u << 16);    // tensor_dim0[15:0] = 8192
    g1[2] = (int)(1u << 16);         // tensor_dim0[31:16]=0, tensor_dim1=1
    g1[3] = (int)(0x2000u << 16);    // tile_dim0 = 8192  (8192*8B = 64 KB)
    g1[4] = 1;                       // tile_dim1 = 1, tile_dim2 = 0
    g1[5] = 0x2000;                  // tensor_dim0_stride = 8192
    g1[6] = 0;
    g1[7] = 0;
    v4i z4 = {0, 0, 0, 0};
#if defined(__clang_major__) && (__clang_major__ >= 23)
    v8i z8 = {0, 0, 0, 0, 0, 0, 0, 0};
    __builtin_amdgcn_tensor_load_to_lds(g0, g1, z4, z4, z8, 0);
#else
    __builtin_amdgcn_tensor_load_to_lds(g0, g1, z4, z4, 0);
#endif
}
#endif

// ---------------------------------------------------------------------------
// Main GEMM. 8 waves/WG, each wave owns a 16-row M-tile, computes 16x64 per
// column chunk. A split hi/lo in registers from global f32 (L2-hot); B from
// TDM-staged LDS, one contiguous 32B load per fragment.
// ---------------------------------------------------------------------------
__global__ __launch_bounds__(WG_THREADS) void deletion_gemm(
    const float* __restrict__ x,
    const int*   __restrict__ mask,
    const bf16*  __restrict__ whiT,
    const bf16*  __restrict__ wloT,
    float*       __restrict__ out) {

    extern __shared__ char smem[];
    bf16* ldsBhi = (bf16*)smem;                 // 64 KB
    bf16* ldsBlo = ldsBhi + LDSB_ELEMS;         // 64 KB

    const int tid  = threadIdx.x;
    const int wave = tid >> 5;
    const int lane = tid & 31;
    const int half = lane >> 4;
    const int l16  = lane & 15;

    const int mBase = blockIdx.x * ROWS_PER_WG + wave * ROWS_PER_WAVE;

    int mrow[8];                                 // C VGPR j -> row mBase+8*half+j
#pragma unroll
    for (int j = 0; j < 8; ++j) mrow[j] = mask[mBase + 8 * half + j];

    const float* xrow = x + (size_t)(mBase + l16) * DIM;

    for (int c = 0; c < NCHUNKS; ++c) {
        const int n0 = c * NCHUNK;

        __syncthreads();                         // prior chunk fully consumed
#if HAVE_TDM
        if (wave == 0) {
            tdm_load_64kb((u32)(uintptr_t)(void*)ldsBhi, whiT + (size_t)n0 * DIM);
            tdm_load_64kb((u32)(uintptr_t)(void*)ldsBlo, wloT + (size_t)n0 * DIM);
#if HAVE_TCNT
            __builtin_amdgcn_s_wait_tensorcnt(0);
#else
            asm volatile("s_wait_tensorcnt 0x0" ::: "memory");
#endif
        }
#else
        {
            const uint4* sh = (const uint4*)(whiT + (size_t)n0 * DIM);
            const uint4* sl = (const uint4*)(wloT + (size_t)n0 * DIM);
            uint4* dh = (uint4*)ldsBhi;
            uint4* dl = (uint4*)ldsBlo;
#pragma unroll 4
            for (int i = tid; i < LDSB_ELEMS / 8; i += WG_THREADS) {
                dh[i] = sh[i];
                dl[i] = sl[i];
            }
        }
#endif
        __syncthreads();

        v8f acc[4];
#pragma unroll
        for (int t = 0; t < 4; ++t) acc[t] = (v8f){0, 0, 0, 0, 0, 0, 0, 0};

        // Prefetch next chunk of W toward L2 while computing this one.
        if (c + 1 < NCHUNKS) {
            const char* ph = (const char*)(whiT + (size_t)(n0 + NCHUNK) * DIM);
            const char* pl = (const char*)(wloT + (size_t)(n0 + NCHUNK) * DIM);
            __builtin_prefetch(ph + tid * 256, 0, 1);
            __builtin_prefetch(pl + tid * 256, 0, 1);
        }

#pragma unroll 2
        for (int k = 0; k < DIM; k += 32) {
            // ---- A fragment: 16 f32 from global, split hi/lo in regs ----
            const float* ap = xrow + k + 16 * half;
            float4 a0 = *(const float4*)(ap);
            float4 a1 = *(const float4*)(ap + 4);
            float4 a2 = *(const float4*)(ap + 8);
            float4 a3 = *(const float4*)(ap + 12);
            float av[16] = {a0.x, a0.y, a0.z, a0.w, a1.x, a1.y, a1.z, a1.w,
                            a2.x, a2.y, a2.z, a2.w, a3.x, a3.y, a3.z, a3.w};
            v16bf ahi, alo;
#pragma unroll
            for (int i = 0; i < 16; ++i) {
                bf16 h = (bf16)av[i];
                ahi[i] = h;
                alo[i] = (bf16)(av[i] - (float)h);
            }

            // ---- Load ALL B fragments first (distinct regs, one ds wait) --
            v16bf bhi[4], blo[4];
#pragma unroll
            for (int t = 0; t < 4; ++t) {
                const int boff = (16 * t + l16) * DIM + k + 16 * half;
                bhi[t] = *(const v16bf*)(ldsBhi + boff);
                blo[t] = *(const v16bf*)(ldsBlo + boff);
            }

            // ---- 12 WMMAs, product-major: consecutive ops hit different D
#pragma unroll
            for (int t = 0; t < 4; ++t)
                acc[t] = __builtin_amdgcn_wmma_f32_16x16x32_bf16(
                    false, ahi, false, bhi[t], (short)0, acc[t], false, false);
#pragma unroll
            for (int t = 0; t < 4; ++t)
                acc[t] = __builtin_amdgcn_wmma_f32_16x16x32_bf16(
                    false, alo, false, bhi[t], (short)0, acc[t], false, false);
#pragma unroll
            for (int t = 0; t < 4; ++t)
                acc[t] = __builtin_amdgcn_wmma_f32_16x16x32_bf16(
                    false, ahi, false, blo[t], (short)0, acc[t], false, false);
        }

        // ---- Store: masked rows -> GEMM result, else bit-exact x ----------
#pragma unroll
        for (int t = 0; t < 4; ++t) {
            const int ng = n0 + 16 * t + l16;
#pragma unroll
            for (int j = 0; j < 8; ++j) {
                const int rg = mBase + 8 * half + j;
                float v = mrow[j] ? acc[t][j] : x[(size_t)rg * DIM + ng];
                out[(size_t)rg * DIM + ng] = v;
            }
        }
    }
}

// ---------------------------------------------------------------------------
extern "C" void kernel_launch(void* const* d_in, const int* in_sizes, int n_in,
                              void* d_out, int out_size, void* d_ws, size_t ws_size,
                              hipStream_t stream) {
    const float* x    = (const float*)d_in[0];
    const int*   mask = (const int*)d_in[1];
    const float* W    = (const float*)d_in[2];
    float*       out  = (float*)d_out;

    bf16* whiT = (bf16*)d_ws;
    bf16* wloT = whiT + (size_t)DIM * DIM;

    split_transpose_w<<<(DIM * DIM) / 256, 256, 0, stream>>>(W, whiT, wloT);

    const int rows = in_sizes[0] / DIM;
    const int grid = rows / ROWS_PER_WG;
    deletion_gemm<<<grid, WG_THREADS, 2 * LDSB_ELEMS * (int)sizeof(bf16), stream>>>(
        x, mask, whiT, wloT, out);
}
